// RecurrentGCN_regression_31937376813749
// MI455X (gfx1250) — matrix operand order repaired and checked by reading, not compiled, not run on hardware
//
#include <hip/hip_runtime.h>

typedef __attribute__((ext_vector_type(16))) _Float16 v16h;
typedef __attribute__((ext_vector_type(8)))  float    v8f;
typedef __attribute__((ext_vector_type(4)))  float    v4f;

#define N_GRAPHS        4096
#define NODES_PER_GRAPH 256
#define F_IN            16
#define HID             32
#define D_CAT           48   // F_IN + HID (row count of W_z / W_h slabs)

#define LOG2E 1.44269504088896340736f

__device__ __forceinline__ float fast_tanh(float x) {
#if __has_builtin(__builtin_amdgcn_tanhf)
  return __builtin_amdgcn_tanhf(x);        // single v_tanh_f32
#else
  float e = __builtin_amdgcn_exp2f(-2.0f * LOG2E * x);
  return __builtin_fmaf(2.0f, __builtin_amdgcn_rcpf(1.0f + e), -1.0f);
#endif
}

// d_ws layout: 4 fp16 B-tiles for V_WMMA_F32_16X16X32_F16, each [32 lanes][16 halves]
//   tile 0: Wz_eff*log2e cols  0..15   tile 1: Wz_eff*log2e cols 16..31  (exp2 domain)
//   tile 2: Wh_eff       cols  0..15   tile 3: Wh_eff       cols 16..31
// B (K=32 x N=16) wave32 layout: n=lane&15; halves 0-7 hold K=8*(lane>>4)+j (real data, K<16);
//                                halves 8-15 hold K=16+8*(lane>>4)+(j-8) (zero pad).
__global__ void dcrnn_prep_kernel(const float* __restrict__ Wz,
                                  const float* __restrict__ Wh,
                                  _Float16* __restrict__ wsB) {
  int idx = blockIdx.x * blockDim.x + threadIdx.x;
  if (idx >= 4 * 32 * 16) return;
  int tile = idx >> 9;          // 0..3
  int rem  = idx & 511;
  int lane = rem >> 4;          // 0..31
  int j    = rem & 15;          // half index within lane
  int k    = ((lane >> 4) << 3) + (j & 7) + ((j >> 3) << 4);  // K = 0..31
  int col  = ((tile & 1) << 4) + (lane & 15);     // output column 0..31
  const float* W = (tile < 2) ? Wz : Wh;          // shape (2,1,48,32) flat
  float v = 0.0f;
  if (k < F_IN) {  // W_eff = W[0,0] + W[1,0], rows 0..15 only (h0 == 0)
    v = W[k * HID + col] + W[D_CAT * HID + k * HID + col];
    if (tile < 2) v *= LOG2E;  // fold log2(e) so Z is already in exp2 domain
  }
  wsB[idx] = (_Float16)v;
}

__global__ __launch_bounds__(512)
void dcrnn_pool_kernel(const float* __restrict__ x,
                       const _Float16* __restrict__ wsB,
                       const float* __restrict__ bz,
                       const float* __restrict__ bh,
                       const float* __restrict__ wlin,
                       const float* __restrict__ blin,
                       float* __restrict__ out) {
  __shared__ float smem[16];
  const int tid  = threadIdx.x;
  const int wave = tid >> 5;          // 16 waves/block = one 256-node graph
  const int lane = tid & 31;
  const int l16  = lane & 15;
  const int grp  = lane >> 4;

  // A operand: 16 nodes x 16 features, zero-padded to K=32 in fp16.
  // Lane layout (ISA 7.12.2): M = lane&15; halves 0-7 = K 8*grp..8*grp+7; halves 8-15 = 0.
  const int node = blockIdx.x * NODES_PER_GRAPH + wave * 16 + l16;
  const float* xr = x + (size_t)node * F_IN + grp * 8;
  v4f x0 = __builtin_nontemporal_load((const v4f*)xr);       // x is streamed once: NT
  v4f x1 = __builtin_nontemporal_load((const v4f*)(xr + 4));

  v16h a = {};
  a[0] = (_Float16)x0[0]; a[1] = (_Float16)x0[1];
  a[2] = (_Float16)x0[2]; a[3] = (_Float16)x0[3];
  a[4] = (_Float16)x1[0]; a[5] = (_Float16)x1[1];
  a[6] = (_Float16)x1[2]; a[7] = (_Float16)x1[3];

  const v16h b0 = *(const v16h*)(wsB + 0 * 512 + lane * 16);
  const v16h b1 = *(const v16h*)(wsB + 1 * 512 + lane * 16);
  const v16h b2 = *(const v16h*)(wsB + 2 * 512 + lane * 16);
  const v16h b3 = *(const v16h*)(wsB + 3 * 512 + lane * 16);

  // Biases ride in as the WMMA C operand (D = A*B + C); per-lane C is a splat
  // of this lane's column bias. Z biases pre-scaled by log2e (exp2 domain).
  const float bzs0 = bz[l16] * LOG2E, bzs1 = bz[l16 + 16] * LOG2E;
  const float bh0  = bh[l16],         bh1  = bh[l16 + 16];
  const float wl0  = wlin[l16],       wl1  = wlin[l16 + 16];

  v8f cz0 = {bzs0, bzs0, bzs0, bzs0, bzs0, bzs0, bzs0, bzs0};
  v8f cz1 = {bzs1, bzs1, bzs1, bzs1, bzs1, bzs1, bzs1, bzs1};
  v8f ch0 = {bh0, bh0, bh0, bh0, bh0, bh0, bh0, bh0};
  v8f ch1 = {bh1, bh1, bh1, bh1, bh1, bh1, bh1, bh1};

  v8f accZ0 = __builtin_amdgcn_wmma_f32_16x16x32_f16(false, a, false, b0, (short)0, cz0, false, false);
  v8f accZ1 = __builtin_amdgcn_wmma_f32_16x16x32_f16(false, a, false, b1, (short)0, cz1, false, false);
  v8f accH0 = __builtin_amdgcn_wmma_f32_16x16x32_f16(false, a, false, b2, (short)0, ch0, false, false);
  v8f accH1 = __builtin_amdgcn_wmma_f32_16x16x32_f16(false, a, false, b3, (short)0, ch1, false, false);

  float local = 0.0f;
#pragma unroll
  for (int r = 0; r < 8; ++r) {
    // 1 - sigmoid(z) = 1/(1 + e^z) = rcp(1 + exp2(z*log2e)); accZ already scaled.
    float i0 = __builtin_amdgcn_rcpf(1.0f + __builtin_amdgcn_exp2f(accZ0[r]));
    float i1 = __builtin_amdgcn_rcpf(1.0f + __builtin_amdgcn_exp2f(accZ1[r]));
    float g0 = i0 * fast_tanh(accH0[r]);   // H = (1-Z)*H_tilde  (h0 == 0)
    float g1 = i1 * fast_tanh(accH1[r]);
    local = __builtin_fmaf(fmaxf(g0, 0.0f), wl0, local);  // relu then dot W_lin
    local = __builtin_fmaf(fmaxf(g1, 0.0f), wl1, local);
  }

  // wave32 all-lane sum
#pragma unroll
  for (int off = 16; off > 0; off >>= 1)
    local += __shfl_xor(local, off, 32);

  if (lane == 0) smem[wave] = local;
  __syncthreads();
  if (tid == 0) {
    float tot = 0.0f;
#pragma unroll
    for (int w = 0; w < 16; ++w) tot += smem[w];
    out[blockIdx.x] = blin[0] + tot * (1.0f / NODES_PER_GRAPH);  // mean pool + bias
  }
}

extern "C" void kernel_launch(void* const* d_in, const int* in_sizes, int n_in,
                              void* d_out, int out_size, void* d_ws, size_t ws_size,
                              hipStream_t stream) {
  (void)in_sizes; (void)n_in; (void)out_size; (void)ws_size;
  const float* x    = (const float*)d_in[0];
  // d_in[1..3]: edge_index / edge_weight / batch — mathematically unused (K=1, contiguous batch)
  const float* Wz   = (const float*)d_in[4];
  const float* bz   = (const float*)d_in[5];
  // d_in[6..7]: W_r / b_r dead (h0 == 0)
  const float* Wh   = (const float*)d_in[8];
  const float* bh   = (const float*)d_in[9];
  const float* wlin = (const float*)d_in[10];
  const float* blin = (const float*)d_in[11];
  _Float16* wsB = (_Float16*)d_ws;  // 4 KB of packed fp16 B tiles

  dcrnn_prep_kernel<<<8, 256, 0, stream>>>(Wz, Wh, wsB);
  dcrnn_pool_kernel<<<N_GRAPHS, 512, 0, stream>>>(x, wsB, bz, bh, wlin, blin, (float*)d_out);
}
